// SignalTransformerClassifier_43946105373132
// MI455X (gfx1250) — compile-verified, matrix operand order checked by reading, not run on hardware
//
#include <hip/hip_runtime.h>

// Problem constants (match reference)
#define B_    4
#define S_    512
#define FEAT_ 16
#define D_    512
#define H_    8
#define L_    6
#define DFF_  2048
#define NC_   12
#define RPH_  64
#define DH_   64
#define BS_   (B_ * S_)

typedef __attribute__((ext_vector_type(16))) __bf16 v16bf;
typedef __attribute__((ext_vector_type(8)))  __bf16 v8bf;
typedef __attribute__((ext_vector_type(8)))  float  v8f;

union FragBF { v16bf v; v8bf h[2]; };

__device__ __forceinline__ unsigned short f2bf(float x) {
  unsigned int u = __float_as_uint(x);
  u += 0x7fffu + ((u >> 16) & 1u);   // round-to-nearest-even
  return (unsigned short)(u >> 16);
}

__device__ __forceinline__ float blk_reduce_sum(float v, float* red) {
  int t = threadIdx.x;
  red[t] = v;
  __syncthreads();
  for (int s = blockDim.x >> 1; s > 0; s >>= 1) {
    if (t < s) red[t] += red[t + s];
    __syncthreads();
  }
  float r = red[0];
  __syncthreads();
  return r;
}

// ---------------- standardize x[...,0], x[...,1] over S ----------------
__global__ __launch_bounds__(512) void standardize_kernel(
    const float* __restrict__ x, float* __restrict__ tstd, float* __restrict__ fstd) {
  __shared__ float red[512];
  int b = blockIdx.x, s = threadIdx.x;
  float tv = x[(b * S_ + s) * FEAT_ + 0];
  float fv = x[(b * S_ + s) * FEAT_ + 1];
  float tm = blk_reduce_sum(tv, red) * (1.0f / S_);
  float tvar = blk_reduce_sum((tv - tm) * (tv - tm), red) * (1.0f / S_);
  tstd[b * S_ + s] = (tv - tm) * rsqrtf(tvar + 1e-6f);
  float fm = blk_reduce_sum(fv, red) * (1.0f / S_);
  float fvar = blk_reduce_sum((fv - fm) * (fv - fm), red) * (1.0f / S_);
  fstd[b * S_ + s] = (fv - fm) * rsqrtf(fvar + 1e-6f);
}

// ---------------- relative position bias MLP: (B,S,S)->(B,H,S,S) ----------------
__global__ __launch_bounds__(256) void relpos_kernel(
    const float* __restrict__ tstd, const float* __restrict__ fstd,
    const float* __restrict__ w1, const float* __restrict__ b1,
    const float* __restrict__ w2, const float* __restrict__ b2,
    float* __restrict__ biasb) {
  __shared__ float sw1[RPH_ * 2], sb1[RPH_], sw2[H_ * RPH_], sb2[H_];
  for (int i = threadIdx.x; i < RPH_ * 2; i += 256) sw1[i] = w1[i];
  for (int i = threadIdx.x; i < RPH_; i += 256) sb1[i] = b1[i];
  for (int i = threadIdx.x; i < H_ * RPH_; i += 256) sw2[i] = w2[i];
  if (threadIdx.x < H_) sb2[threadIdx.x] = b2[threadIdx.x];
  __syncthreads();
  int idx = blockIdx.x * 256 + threadIdx.x;     // over B*S*S
  int j = idx & (S_ - 1);
  int i = (idx >> 9) & (S_ - 1);
  int b = idx >> 18;
  float dt = tstd[b * S_ + i] - tstd[b * S_ + j];
  float df = fstd[b * S_ + i] - fstd[b * S_ + j];
  float acc[H_];
#pragma unroll
  for (int h = 0; h < H_; ++h) acc[h] = sb2[h];
  for (int r = 0; r < RPH_; ++r) {
    float hd = fmaxf(fmaf(sw1[r * 2], dt, fmaf(sw1[r * 2 + 1], df, sb1[r])), 0.0f);
#pragma unroll
    for (int h = 0; h < H_; ++h) acc[h] = fmaf(sw2[h * RPH_ + r], hd, acc[h]);
  }
#pragma unroll
  for (int h = 0; h < H_; ++h) {
    float v = fminf(fmaxf(acc[h], -5.0f), 5.0f);
    biasb[(((size_t)b * H_ + h) * S_ + i) * S_ + j] = v;
  }
}

// ---------------- embed: x(BS,16) @ W^T(16,512) + b -> h fp32 + bf16 ----------------
__global__ __launch_bounds__(256) void embed_kernel(
    const float* __restrict__ x, const float* __restrict__ w, const float* __restrict__ b,
    float* __restrict__ h, unsigned short* __restrict__ hbf) {
  int idx = blockIdx.x * 256 + threadIdx.x;   // BS*D
  int d = idx & (D_ - 1);
  int bs = idx >> 9;
  const float* xr = x + bs * FEAT_;
  const float* wr = w + d * FEAT_;
  float acc = b[d];
#pragma unroll
  for (int f = 0; f < FEAT_; ++f) acc = fmaf(xr[f], wr[f], acc);
  h[idx] = acc;
  hbf[idx] = f2bf(acc);
}

// ---------------- fp32 -> bf16 bulk convert ----------------
__global__ __launch_bounds__(256) void cvt_kernel(
    const float* __restrict__ in, unsigned short* __restrict__ out, int n) {
  int i = blockIdx.x * 256 + threadIdx.x;
  if (i < n) out[i] = f2bf(in[i]);
}

// ---------------- V transpose: qkv_bf (B,S,3D) -> vT (B,H,DH,S) bf16 ----------------
__global__ __launch_bounds__(256) void vtrans_kernel(
    const unsigned short* __restrict__ qkvbf, unsigned short* __restrict__ vT) {
  int idx = blockIdx.x * 256 + threadIdx.x;   // B*H*DH*S
  int j = idx & (S_ - 1);
  int d = (idx >> 9) & (DH_ - 1);
  int bh = idx >> 15;
  int h = bh & (H_ - 1);
  int b = bh >> 3;
  vT[idx] = qkvbf[((size_t)(b * S_ + j)) * (3 * D_) + 2 * D_ + h * DH_ + d];
}

// ---------------- WMMA bf16 GEMM: C[M,N] = A[M,K] @ W[N,K]^T + bias ----------------
// Block = 8 waves computing a 256(M) x 64(N) macro-tile.  The 64x32 bf16 B panel
// is staged in LDS (double-buffered, one __syncthreads per K-step) and shared by
// all 8 waves -> 8x less L2 weight traffic.  Each wave computes 32x64 using two
// A fragments and 8 v_wmma_f32_16x16x32_bf16 per K-step (B fragment reused).
// Fragment layouts per CDNA5 ISA 7.12.2.  Requires M % 256 == 0, N % 64 == 0,
// K % 32 == 0.
__global__ __launch_bounds__(256) void gemm_wmma_kernel(
    const unsigned short* __restrict__ A,   // M x K bf16 (row-major)
    const unsigned short* __restrict__ W,   // N x K bf16 (row-major)
    const float* __restrict__ bias,         // N (nullable)
    float* __restrict__ C,                  // M x N fp32 (nullable)
    unsigned short* __restrict__ Cbf,       // M x N bf16 (nullable)
    int M, int N, int K, int relu) {
  __shared__ __align__(16) unsigned short sB[2][64 * 32];

  const int lane = threadIdx.x & 31;
  const int wave = threadIdx.x >> 5;
  const int tilesN = N >> 6;
  const int bm = blockIdx.x / tilesN;
  const int bn = blockIdx.x % tilesN;
  const int m0 = bm * 256 + wave * 32;
  const int n0 = bn * 64;
  const int r = lane & 15;
  const int half = lane >> 4;

  // B panel fill assignment: one b128 per thread per K-step
  const int frow = threadIdx.x >> 2;          // 0..63  (n within panel)
  const int fchk = (threadIdx.x & 3) * 8;     // 0,8,16,24 (k within panel)
  const unsigned short* fsrc = W + (size_t)(n0 + frow) * K + fchk;

  const unsigned short* arow0 = A + (size_t)(m0 + r) * K + half * 8;
  const unsigned short* arow1 = arow0 + (size_t)16 * K;

  // prologue: stage B panel for k=0
  *(v8bf*)(&sB[0][frow * 32 + fchk]) = *(const v8bf*)(fsrc);
  __syncthreads();

  v8f acc[8];
#pragma unroll
  for (int t = 0; t < 8; ++t) acc[t] = v8f{};

  for (int k = 0; k < K; k += 32) {
    const int p = (k >> 5) & 1;
    if (k + 32 < K) {   // stage next B panel into the other buffer
      *(v8bf*)(&sB[1 - p][frow * 32 + fchk]) = *(const v8bf*)(fsrc + k + 32);
      __builtin_prefetch(fsrc + k + 64, 0, 1);
    }
    FragBF a0, a1;
    a0.h[0] = *(const v8bf*)(arow0 + k);
    a0.h[1] = *(const v8bf*)(arow0 + k + 16);
    a1.h[0] = *(const v8bf*)(arow1 + k);
    a1.h[1] = *(const v8bf*)(arow1 + k + 16);
    __builtin_prefetch(arow0 + k + 32, 0, 1);
    const unsigned short* bp = sB[p];
#pragma unroll
    for (int t = 0; t < 4; ++t) {
      FragBF b;
      const int rowp = t * 16 + r;
      b.h[0] = *(const v8bf*)(bp + rowp * 32 + half * 8);
      b.h[1] = *(const v8bf*)(bp + rowp * 32 + 16 + half * 8);
      acc[t]     = __builtin_amdgcn_wmma_f32_16x16x32_bf16(false, a0.v, false, b.v, (short)0, acc[t],     false, false);
      acc[4 + t] = __builtin_amdgcn_wmma_f32_16x16x32_bf16(false, a1.v, false, b.v, (short)0, acc[4 + t], false, false);
    }
    __syncthreads();  // reads of sB[p] done; fill of sB[1-p] visible
  }

#pragma unroll
  for (int half16 = 0; half16 < 2; ++half16) {     // 0: rows m0.., 1: rows m0+16..
#pragma unroll
    for (int t = 0; t < 4; ++t) {
      int n = n0 + t * 16 + r;
      float bv = bias ? bias[n] : 0.0f;
#pragma unroll
      for (int i = 0; i < 8; ++i) {
        int m = m0 + half16 * 16 + half * 8 + i;
        float v = acc[half16 * 4 + t][i] + bv;
        if (relu) v = fmaxf(v, 0.0f);
        size_t o = (size_t)m * N + n;
        if (C)   C[o] = v;
        if (Cbf) Cbf[o] = f2bf(v);
      }
    }
  }
}

// ---------------- WMMA flash attention ----------------
// One wave (32 threads) per (b, h, 16-query tile).
// Phase 1: scores = 0.125 * Q K^T via 32 j-tiles x 2 WMMA -> fp32 LDS (16x512).
// Phase 2: fp32 softmax per row (bias added during scan; contiguous bias reads);
//          unnormalized exp stored bf16 to LDS P panel; 1/sum kept per row.
// Phase 3: O = P V via vT panels, 16 K-steps x 4 WMMA; normalize in epilogue.
__global__ __launch_bounds__(32) void attention_wmma_kernel(
    const unsigned short* __restrict__ qkvbf, // (B,S,3D) bf16
    const unsigned short* __restrict__ vT,    // (B,H,DH,S) bf16
    const float* __restrict__ biasb,          // (B,H,S,S) fp32
    unsigned short* __restrict__ obf) {       // (B,S,D) bf16
  __shared__ __align__(16) float sScore[16 * 512];        // 32KB
  __shared__ __align__(16) unsigned short sP[16 * 512];   // 16KB
  __shared__ float sRow[16];                              // 1/sum per row

  const int lane = threadIdx.x;
  const int r = lane & 15;
  const int half = lane >> 4;

  const int qt = blockIdx.x & 31;           // query tile
  const int bh = blockIdx.x >> 5;           // 0..31
  const int h  = bh & (H_ - 1);
  const int b  = bh >> 3;
  const int q0 = qt * 16;

  const unsigned short* qbase = qkvbf + ((size_t)(b * S_ + q0)) * (3 * D_) + h * DH_;
  const unsigned short* kbase = qkvbf + ((size_t)b * S_) * (3 * D_) + D_ + h * DH_;
  const unsigned short* vtb   = vT + (size_t)bh * DH_ * S_;
  const float* biasq = biasb + ((size_t)bh * S_ + q0) * S_;

  // Q A-fragments: row m = r, d chunks; two K-steps (d 0..31 and 32..63)
  FragBF aq0, aq1;
  {
    const unsigned short* qrow = qbase + (size_t)r * (3 * D_) + half * 8;
    aq0.h[0] = *(const v8bf*)(qrow);
    aq0.h[1] = *(const v8bf*)(qrow + 16);
    aq1.h[0] = *(const v8bf*)(qrow + 32);
    aq1.h[1] = *(const v8bf*)(qrow + 48);
  }

  // ---- phase 1: scaled scores -> LDS ----
  for (int jt = 0; jt < 32; ++jt) {
    const unsigned short* krow = kbase + (size_t)(jt * 16 + r) * (3 * D_) + half * 8;
    FragBF bk0, bk1;
    bk0.h[0] = *(const v8bf*)(krow);
    bk0.h[1] = *(const v8bf*)(krow + 16);
    bk1.h[0] = *(const v8bf*)(krow + 32);
    bk1.h[1] = *(const v8bf*)(krow + 48);
    v8f c = {};
    c = __builtin_amdgcn_wmma_f32_16x16x32_bf16(false, aq0.v, false, bk0.v, (short)0, c, false, false);
    c = __builtin_amdgcn_wmma_f32_16x16x32_bf16(false, aq1.v, false, bk1.v, (short)0, c, false, false);
    const int j = jt * 16 + r;
#pragma unroll
    for (int i = 0; i < 8; ++i) {
      int m = half * 8 + i;
      sScore[m * 512 + j] = c[i] * 0.125f;   // 1/sqrt(DH)
    }
  }
  __syncthreads();

  // ---- phase 2: softmax (row m = r; lane halves split the j range) ----
  {
    const int m = r;
    const int j0 = half * 256;
    float* srow = sScore + m * 512;
    const float* brow = biasq + (size_t)m * S_;
    float mx = -3.0e38f;
    for (int j = j0; j < j0 + 256; ++j) {
      float v = srow[j] + brow[j];
      srow[j] = v;
      mx = fmaxf(mx, v);
    }
    mx = fmaxf(mx, __shfl_xor(mx, 16));
    float sum = 0.0f;
    for (int j = j0; j < j0 + 256; ++j) {
      float p = __expf(srow[j] - mx);
      sP[m * 512 + j] = f2bf(p);
      sum += p;
    }
    sum += __shfl_xor(sum, 16);
    if (half == 0) sRow[m] = 1.0f / sum;
  }
  __syncthreads();

  // ---- phase 3: O = P @ V  (vT rows are K-contiguous) ----
  v8f oacc[4];
#pragma unroll
  for (int t = 0; t < 4; ++t) oacc[t] = v8f{};
  for (int k = 0; k < 512; k += 32) {
    FragBF ap;
    const unsigned short* prow = sP + (size_t)r * 512 + k + half * 8;
    ap.h[0] = *(const v8bf*)(prow);
    ap.h[1] = *(const v8bf*)(prow + 16);
#pragma unroll
    for (int t = 0; t < 4; ++t) {
      const unsigned short* vrow = vtb + (size_t)(t * 16 + r) * S_ + k + half * 8;
      FragBF bv;
      bv.h[0] = *(const v8bf*)(vrow);
      bv.h[1] = *(const v8bf*)(vrow + 16);
      oacc[t] = __builtin_amdgcn_wmma_f32_16x16x32_bf16(false, ap.v, false, bv.v, (short)0, oacc[t], false, false);
    }
  }

  // ---- epilogue: normalize rows, store bf16 ----
#pragma unroll
  for (int t = 0; t < 4; ++t) {
    int d = t * 16 + r;
#pragma unroll
    for (int i = 0; i < 8; ++i) {
      int m = half * 8 + i;
      float v = oacc[t][i] * sRow[m];
      obf[((size_t)(b * S_ + q0 + m)) * D_ + h * DH_ + d] = f2bf(v);
    }
  }
}

// ---------------- residual add + LayerNorm, emits fp32 + bf16 ----------------
__global__ __launch_bounds__(256) void add_ln_kernel(
    float* __restrict__ h, const float* __restrict__ dlt,
    const float* __restrict__ sc, const float* __restrict__ bt,
    unsigned short* __restrict__ hbf) {
  __shared__ float red[256];
  int row = blockIdx.x;                 // B*S rows
  int t = threadIdx.x;
  size_t base = (size_t)row * D_;
  float a0 = h[base + t] + dlt[base + t];
  float a1 = h[base + t + 256] + dlt[base + t + 256];
  float mean = blk_reduce_sum(a0 + a1, red) * (1.0f / D_);
  float d0 = a0 - mean, d1 = a1 - mean;
  float var = blk_reduce_sum(d0 * d0 + d1 * d1, red) * (1.0f / D_);
  float inv = rsqrtf(var + 1e-5f);
  float o0 = fmaf(d0 * inv, sc[t], bt[t]);
  float o1 = fmaf(d1 * inv, sc[t + 256], bt[t + 256]);
  h[base + t] = o0;        hbf[base + t] = f2bf(o0);
  h[base + t + 256] = o1;  hbf[base + t + 256] = f2bf(o1);
}

// ---------------- mean pool over S ----------------
__global__ __launch_bounds__(256) void pool_kernel(
    const float* __restrict__ h, float* __restrict__ rep) {
  int idx = blockIdx.x * 256 + threadIdx.x;   // B*D
  int d = idx & (D_ - 1);
  int b = idx >> 9;
  float acc = 0.0f;
  for (int s = 0; s < S_; ++s) acc += h[((size_t)(b * S_ + s)) * D_ + d];
  rep[idx] = acc * (1.0f / S_);
}

// ---------------- classifier head ----------------
__global__ __launch_bounds__(64) void fc_kernel(
    const float* __restrict__ rep, const float* __restrict__ w,
    const float* __restrict__ b, float* __restrict__ out) {
  int t = threadIdx.x;
  if (t >= B_ * NC_) return;
  int bb = t / NC_, c = t % NC_;
  float acc = b[c];
  for (int d = 0; d < D_; ++d) acc = fmaf(rep[bb * D_ + d], w[c * D_ + d], acc);
  out[t] = acc;
}

extern "C" void kernel_launch(void* const* d_in, const int* in_sizes, int n_in,
                              void* d_out, int out_size, void* d_ws, size_t ws_size,
                              hipStream_t stream) {
  (void)in_sizes; (void)n_in; (void)out_size; (void)ws_size;
  const float* x          = (const float*)d_in[0];
  const float* relpos_w1  = (const float*)d_in[1];
  const float* relpos_b1  = (const float*)d_in[2];
  const float* relpos_w2  = (const float*)d_in[3];
  const float* relpos_b2  = (const float*)d_in[4];
  const float* embed_w    = (const float*)d_in[5];
  const float* embed_b    = (const float*)d_in[6];
  const float* attn_in_w  = (const float*)d_in[7];
  const float* attn_in_b  = (const float*)d_in[8];
  const float* attn_out_w = (const float*)d_in[9];
  const float* attn_out_b = (const float*)d_in[10];
  const float* ff_w1      = (const float*)d_in[11];
  const float* ff_b1      = (const float*)d_in[12];
  const float* ff_w2      = (const float*)d_in[13];
  const float* ff_b2      = (const float*)d_in[14];
  const float* ln1_s      = (const float*)d_in[15];
  const float* ln1_b      = (const float*)d_in[16];
  const float* ln2_s      = (const float*)d_in[17];
  const float* ln2_b      = (const float*)d_in[18];
  const float* fc_w       = (const float*)d_in[19];
  const float* fc_b       = (const float*)d_in[20];

  char* wsb = (char*)d_ws;
  size_t off = 0;
  auto carve = [&](size_t bytes) -> void* {
    off = (off + 255) & ~(size_t)255;
    void* p = wsb + off;
    off += bytes;
    return p;
  };
  float* tstd            = (float*)carve((size_t)B_ * S_ * 4);
  float* fstd            = (float*)carve((size_t)B_ * S_ * 4);
  float* biasb           = (float*)carve((size_t)B_ * H_ * S_ * S_ * 4);
  float* h               = (float*)carve((size_t)BS_ * D_ * 4);
  unsigned short* hbf    = (unsigned short*)carve((size_t)BS_ * D_ * 2);
  unsigned short* qkvbf  = (unsigned short*)carve((size_t)BS_ * 3 * D_ * 2);
  unsigned short* vTb    = (unsigned short*)carve((size_t)BS_ * D_ * 2);
  unsigned short* obf    = (unsigned short*)carve((size_t)BS_ * D_ * 2);
  float* tmp             = (float*)carve((size_t)BS_ * D_ * 4);
  unsigned short* ffhbf  = (unsigned short*)carve((size_t)BS_ * DFF_ * 2);
  float* rep             = (float*)carve((size_t)B_ * D_ * 4);
  unsigned short* winbf  = (unsigned short*)carve((size_t)L_ * 3 * D_ * D_ * 2);
  unsigned short* woutbf = (unsigned short*)carve((size_t)L_ * D_ * D_ * 2);
  unsigned short* w1bf   = (unsigned short*)carve((size_t)L_ * DFF_ * D_ * 2);
  unsigned short* w2bf   = (unsigned short*)carve((size_t)L_ * D_ * DFF_ * 2);

  standardize_kernel<<<B_, 512, 0, stream>>>(x, tstd, fstd);
  relpos_kernel<<<(B_ * S_ * S_) / 256, 256, 0, stream>>>(
      tstd, fstd, relpos_w1, relpos_b1, relpos_w2, relpos_b2, biasb);
  embed_kernel<<<(BS_ * D_) / 256, 256, 0, stream>>>(x, embed_w, embed_b, h, hbf);

  int nwin = L_ * 3 * D_ * D_;
  cvt_kernel<<<(nwin + 255) / 256, 256, 0, stream>>>(attn_in_w, winbf, nwin);
  int nwout = L_ * D_ * D_;
  cvt_kernel<<<(nwout + 255) / 256, 256, 0, stream>>>(attn_out_w, woutbf, nwout);
  int nw1 = L_ * DFF_ * D_;
  cvt_kernel<<<(nw1 + 255) / 256, 256, 0, stream>>>(ff_w1, w1bf, nw1);
  int nw2 = L_ * D_ * DFF_;
  cvt_kernel<<<(nw2 + 255) / 256, 256, 0, stream>>>(ff_w2, w2bf, nw2);

  for (int l = 0; l < L_; ++l) {
    {   // qkv (bf16 out) = h @ Win^T + b : M=2048 N=1536 K=512
      int blocks = (BS_ / 256) * ((3 * D_) / 64);
      gemm_wmma_kernel<<<blocks, 256, 0, stream>>>(
          hbf, winbf + (size_t)l * 3 * D_ * D_, attn_in_b + l * 3 * D_,
          nullptr, qkvbf, BS_, 3 * D_, D_, 0);
    }
    vtrans_kernel<<<(BS_ * D_) / 256, 256, 0, stream>>>(qkvbf, vTb);
    attention_wmma_kernel<<<B_ * H_ * (S_ / 16), 32, 0, stream>>>(qkvbf, vTb, biasb, obf);
    {   // out-proj: M=2048 N=512 K=512
      int blocks = (BS_ / 256) * (D_ / 64);
      gemm_wmma_kernel<<<blocks, 256, 0, stream>>>(
          obf, woutbf + (size_t)l * D_ * D_, attn_out_b + l * D_,
          tmp, nullptr, BS_, D_, D_, 0);
    }
    add_ln_kernel<<<BS_, 256, 0, stream>>>(h, tmp, ln1_s + l * D_, ln1_b + l * D_, hbf);
    {   // ff1 (ReLU, bf16 out): M=2048 N=2048 K=512
      int blocks = (BS_ / 256) * (DFF_ / 64);
      gemm_wmma_kernel<<<blocks, 256, 0, stream>>>(
          hbf, w1bf + (size_t)l * DFF_ * D_, ff_b1 + l * DFF_,
          nullptr, ffhbf, BS_, DFF_, D_, 1);
    }
    {   // ff2: M=2048 N=512 K=2048
      int blocks = (BS_ / 256) * (D_ / 64);
      gemm_wmma_kernel<<<blocks, 256, 0, stream>>>(
          ffhbf, w2bf + (size_t)l * D_ * DFF_, ff_b2 + l * D_,
          tmp, nullptr, BS_, D_, DFF_, 0);
    }
    add_ln_kernel<<<BS_, 256, 0, stream>>>(h, tmp, ln2_s + l * D_, ln2_b + l * D_, hbf);
  }

  pool_kernel<<<(B_ * D_) / 256, 256, 0, stream>>>(h, rep);
  fc_kernel<<<1, 64, 0, stream>>>(rep, fc_w, fc_b, (float*)d_out);
}
